// BinarizedLinear_9637906612611
// MI455X (gfx1250) — matrix operand order repaired and checked by reading, not compiled, not run on hardware
//
#include <hip/hip_runtime.h>

// ---------------------------------------------------------------------------
// BinarizedLinear on MI455X (gfx1250):
//   pass 1: X (f32) -> bf16 in workspace, W (f32) -> sign(W) as bf16 (+-1/0)
//   pass 2: tiled WMMA GEMM  C = Xbf @ signWbf^T + bias   (f32 accumulate)
//           global->LDS staging via gfx1250 async-copy (ASYNCcnt), no VGPR
//           data staging, double-buffered LDS.
// ---------------------------------------------------------------------------

typedef __attribute__((ext_vector_type(16))) __bf16 v16bf;
typedef __attribute__((ext_vector_type(8)))  __bf16 v8bf;
typedef __attribute__((ext_vector_type(8)))  float  v8f;

static __device__ __forceinline__ unsigned short f32_to_bf16_rne(float f) {
  unsigned u = __float_as_uint(f);
  u += 0x7FFFu + ((u >> 16) & 1u);   // round-to-nearest-even
  return (unsigned short)(u >> 16);
}

// ---- pass 1a: x fp32 -> bf16 (8 elements / thread) ------------------------
__global__ __launch_bounds__(256) void cvt_x_bf16(const float* __restrict__ x,
                                                  unsigned short* __restrict__ o,
                                                  int n8) {
  int g = blockIdx.x * 256 + threadIdx.x;
  if (g >= n8) return;
  const float4* xp = (const float4*)x;
  float4 a = xp[2 * g + 0];
  float4 b = xp[2 * g + 1];
  union { unsigned short us[8]; uint4 v; } u;
  u.us[0] = f32_to_bf16_rne(a.x); u.us[1] = f32_to_bf16_rne(a.y);
  u.us[2] = f32_to_bf16_rne(a.z); u.us[3] = f32_to_bf16_rne(a.w);
  u.us[4] = f32_to_bf16_rne(b.x); u.us[5] = f32_to_bf16_rne(b.y);
  u.us[6] = f32_to_bf16_rne(b.z); u.us[7] = f32_to_bf16_rne(b.w);
  ((uint4*)o)[g] = u.v;
}

// ---- pass 1b: w fp32 -> sign(w) as bf16 (+1 / -1 / 0) ---------------------
static __device__ __forceinline__ unsigned short sign_bf16(float f) {
  return f > 0.0f ? (unsigned short)0x3F80u
                  : (f < 0.0f ? (unsigned short)0xBF80u : (unsigned short)0u);
}

__global__ __launch_bounds__(256) void cvt_w_sign_bf16(const float* __restrict__ w,
                                                       unsigned short* __restrict__ o,
                                                       int n8) {
  int g = blockIdx.x * 256 + threadIdx.x;
  if (g >= n8) return;
  const float4* wp = (const float4*)w;
  float4 a = wp[2 * g + 0];
  float4 b = wp[2 * g + 1];
  union { unsigned short us[8]; uint4 v; } u;
  u.us[0] = sign_bf16(a.x); u.us[1] = sign_bf16(a.y);
  u.us[2] = sign_bf16(a.z); u.us[3] = sign_bf16(a.w);
  u.us[4] = sign_bf16(b.x); u.us[5] = sign_bf16(b.y);
  u.us[6] = sign_bf16(b.z); u.us[7] = sign_bf16(b.w);
  ((uint4*)o)[g] = u.v;
}

// ---- gfx1250 async global->LDS copy (16B per lane), tracked by ASYNCcnt ---
static __device__ __forceinline__ void async_copy_b128(unsigned lds_addr,
                                                       const void* gaddr) {
  asm volatile("global_load_async_to_lds_b128 %0, %1, off"
               :
               : "v"(lds_addr), "v"((unsigned long long)(size_t)gaddr)
               : "memory");
}

static __device__ __forceinline__ void wait_async0() {
  asm volatile("s_wait_asynccnt 0x0" ::: "memory");
}

// ---- pass 2: WMMA GEMM ----------------------------------------------------
// Block: 256 threads (8 waves, wave32). Tile 128(M) x 128(N), K-step 64.
// Wave grid 4(M) x 2(N): each wave -> 32x64 = 2x4 WMMA accumulators.
// LDS double-buffered, filled by async copies; one barrier per K-stage.

#define TILE_M 128
#define TILE_N 128
#define TILE_K 64
#define LDSS   72   // row stride in halves: 64 + 8 pad (144 B, 16B-aligned)

__global__ __launch_bounds__(256)
void bgemm_wmma_bf16(const unsigned short* __restrict__ A,   // [M,K] bf16 bits
                     const unsigned short* __restrict__ B,   // [N,K] bf16 bits
                     const float* __restrict__ bias,         // [N]
                     float* __restrict__ C,                  // [M,N]
                     int M, int N, int K) {
  __shared__ unsigned short ldsA[2][TILE_M * LDSS];
  __shared__ unsigned short ldsB[2][TILE_N * LDSS];

  const int tid   = threadIdx.x;
  const int lane  = tid & 31;
  const int wave  = tid >> 5;
  const int waveM = wave & 3;     // 0..3  (32 rows each)
  const int waveN = wave >> 2;    // 0..1  (64 cols each)
  const int l16   = lane & 15;
  const int sel   = lane >> 4;    // half-wave selector for fragment layout

  const int blockM = blockIdx.y * TILE_M;
  const int blockN = blockIdx.x * TILE_N;

  // staging decomposition: tile = 128 rows x 8 groups of 8 halves (16B each)
  // 1024 async b128 moves / 256 threads = 4 rounds per operand
  const int srow = tid >> 3;        // base row for round 0 (0..31)
  const int skg  = tid & 7;         // 16B group within row

  // per-round global base pointers (halves) and LDS byte addresses (buffer 0)
  const unsigned short* gA[4];
  const unsigned short* gB[4];
  unsigned lA[4], lB[4];
  const unsigned ldsA0 = (unsigned)(size_t)&ldsA[0][0];
  const unsigned ldsB0 = (unsigned)(size_t)&ldsB[0][0];
#pragma unroll
  for (int r = 0; r < 4; ++r) {
    const int row = srow + r * 32;
    gA[r] = A + (size_t)(blockM + row) * K + skg * 8;
    gB[r] = B + (size_t)(blockN + row) * K + skg * 8;
    const unsigned off = (unsigned)(row * LDSS + skg * 8) * 2u;
    lA[r] = ldsA0 + off;
    lB[r] = ldsB0 + off;
  }
  const unsigned bufBytes = (unsigned)(TILE_M * LDSS * 2);  // 18432

  v8f acc[2][4] = {};
  const int nStages = K / TILE_K;

  // ---- prime stage 0 (async, no data VGPRs) ----
#pragma unroll
  for (int r = 0; r < 4; ++r) {
    async_copy_b128(lA[r], gA[r]);
    async_copy_b128(lB[r], gB[r]);
  }

  for (int s = 0; s < nStages; ++s) {
    wait_async0();      // my async copies for buf (s&1) have landed in LDS
    __syncthreads();    // everyone's have; buf (s&1) is readable

    // ---- issue async copies for stage s+1 into the other buffer ----
    if (s + 1 < nStages) {
      const int kb = (s + 1) * TILE_K;              // halves
      const unsigned bufOff = ((s + 1) & 1) * bufBytes;
#pragma unroll
      for (int r = 0; r < 4; ++r) {
        async_copy_b128(lA[r] + bufOff, gA[r] + kb);
        async_copy_b128(lB[r] + bufOff, gB[r] + kb);
      }
    }

    const int cur = s & 1;

    // ---- compute: two K=32 chunks, 8 WMMA each ----
#pragma unroll
    for (int kc = 0; kc < 2; ++kc) {
      const int kk = kc * 32;

      v16bf af[2];
#pragma unroll
      for (int mt = 0; mt < 2; ++mt) {
        const int row = waveM * 32 + mt * 16 + l16;
        const unsigned short* p = &ldsA[cur][row * LDSS + kk + sel * 8];
        ((v8bf*)&af[mt])[0] = *(const v8bf*)(p);        // K 0..7   / 8..15
        ((v8bf*)&af[mt])[1] = *(const v8bf*)(p + 16);   // K 16..23 / 24..31
      }

      v16bf bfm[4];
#pragma unroll
      for (int nt = 0; nt < 4; ++nt) {
        const int col = waveN * 64 + nt * 16 + l16;
        const unsigned short* p = &ldsB[cur][col * LDSS + kk + sel * 8];
        ((v8bf*)&bfm[nt])[0] = *(const v8bf*)(p);       // K 0..7   / 8..15
        ((v8bf*)&bfm[nt])[1] = *(const v8bf*)(p + 16);  // K 16..23 / 24..31
      }

#pragma unroll
      for (int mt = 0; mt < 2; ++mt)
#pragma unroll
        for (int nt = 0; nt < 4; ++nt)
          acc[mt][nt] = __builtin_amdgcn_wmma_f32_16x16x32_bf16(
              false, af[mt], false, bfm[nt], (short)0, acc[mt][nt],
              false, false);
    }
  }

  // ---- epilogue: C/D layout -> global, add bias ----
  // VGPR j of a 16x16 f32 D tile: lanes 0-15 -> M=j, lanes 16-31 -> M=j+8.
#pragma unroll
  for (int mt = 0; mt < 2; ++mt) {
#pragma unroll
    for (int nt = 0; nt < 4; ++nt) {
      const int col = blockN + waveN * 64 + nt * 16 + l16;
      const float bv = bias[col];
      const int rowBase = blockM + waveM * 32 + mt * 16 + sel * 8;
#pragma unroll
      for (int j = 0; j < 8; ++j)
        C[(size_t)(rowBase + j) * N + col] = acc[mt][nt][j] + bv;
    }
  }
}

// ---------------------------------------------------------------------------
extern "C" void kernel_launch(void* const* d_in, const int* in_sizes, int n_in,
                              void* d_out, int out_size, void* d_ws, size_t ws_size,
                              hipStream_t stream) {
  const float* x    = (const float*)d_in[0];   // [tokens, inF]
  const float* w    = (const float*)d_in[1];   // [outF, inF]
  const float* bias = (const float*)d_in[2];   // [outF]
  float* out        = (float*)d_out;

  const int x_n  = in_sizes[0];
  const int w_n  = in_sizes[1];
  const int outF = in_sizes[2];
  const int inF  = w_n / outF;
  const int tokens = x_n / inF;

  unsigned short* Abf = (unsigned short*)d_ws;       // tokens*inF bf16
  unsigned short* Bbf = Abf + (size_t)x_n;           // outF*inF bf16
  (void)ws_size; (void)n_in; (void)out_size;

  // pass 1: conversions (8 elems/thread)
  {
    int n8 = x_n / 8;
    cvt_x_bf16<<<(n8 + 255) / 256, 256, 0, stream>>>(x, Abf, n8);
  }
  {
    int n8 = w_n / 8;
    cvt_w_sign_bf16<<<(n8 + 255) / 256, 256, 0, stream>>>(w, Bbf, n8);
  }

  // pass 2: WMMA GEMM (dims are multiples of tile sizes for this workload)
  dim3 grid(outF / TILE_N, tokens / TILE_M);
  bgemm_wmma_bf16<<<grid, 256, 0, stream>>>(Abf, Bbf, bias, out,
                                            tokens, outF, inF);
}
